// SimPGCN_88347477279355
// MI455X (gfx1250) — compile-verified, matrix operand order checked by reading, not run on hardware
//
#include <hip/hip_runtime.h>
#include <math.h>

typedef __attribute__((ext_vector_type(16))) _Float16 v16h;
typedef __attribute__((ext_vector_type(8)))  float    v8f;

#define SIMP_GAMMA 0.1f

// ---------------------------------------------------------------------------
// Fused GEMM (x @ W via v_wmma_f32_16x16x32_f16) + per-row gate dot products:
//   s[row]  = sigmoid(x[row,:] . sc + sb)
//   dk[row] = x[row,:] . dkw + db
// Block = 4 waves (128 threads). Each wave owns TWO 16x16 output tiles
// (32 rows) sharing one B fragment -> 2 back-to-back v_wmma per K-step.
//   layer 1: M_TILES=1, N_TILES=4 -> 32 rows x 64 cols per block
//   layer 2: M_TILES=4, N_TILES=1 -> 128 rows x 16 cols per block
// ---------------------------------------------------------------------------
template<int K_DIM, int N_DIM, int M_TILES, int N_TILES>
__global__ __launch_bounds__(32 * M_TILES * N_TILES)
void gemm_gate_wmma(const float* __restrict__ X,     // [n_nodes, K_DIM]
                    const float* __restrict__ Wm,    // [K_DIM, N_DIM]
                    const float* __restrict__ sc,    // [K_DIM]
                    const float* __restrict__ sb,    // [1]
                    const float* __restrict__ dkw,   // [K_DIM]
                    const float* __restrict__ dkb,   // [1]
                    float* __restrict__ XW,          // [n_nodes, N_DIM]
                    float* __restrict__ s_out,       // [n_nodes]
                    float* __restrict__ dk_out,      // [n_nodes]
                    int n_nodes)
{
    constexpr int NW       = M_TILES * N_TILES;   // waves per block (=4)
    constexpr int ROWS     = M_TILES * 32;        // 2 m-tiles per wave
    constexpr int NTHREADS = 32 * NW;
    constexpr int KSTEPS   = (K_DIM + 31) / 32;

    __shared__ float lA[ROWS][34];     // f32 A tile; stride 34 -> conflict-free b64 reads
    __shared__ float lB[N_DIM][34];    // B tile transposed [col][k]

    const int tid    = threadIdx.x;
    const int lane   = tid & 31;
    const int wv     = tid >> 5;
    const int mt     = wv / N_TILES;
    const int nt     = wv % N_TILES;
    const int laneLo = lane & 15;
    const int laneHi = lane >> 4;      // 0 or 1
    const int blockRow = blockIdx.x * ROWS;
    const int rbase  = mt * 32;
    const bool fullRows = (blockRow + ROWS) <= n_nodes;

    v8f acc0 = {}, acc1 = {};
    float gs = 0.f, gd = 0.f;

    for (int ks = 0; ks < KSTEPS; ++ks) {
        const int k0 = ks * 32;
        const bool fullK = (k0 + 32) <= K_DIM;

        // ---- stage A tile (ROWS x 32, f32) ----
        if (fullRows && fullK) {
            // fast path: vectorized b128 loads, 8 float4 per row
            #pragma unroll
            for (int i = tid; i < ROWS * 8; i += NTHREADS) {
                int r = i >> 3, q = i & 7;
                float4 v = *(const float4*)&X[(size_t)(blockRow + r) * K_DIM + k0 + q * 4];
                lA[r][q * 4 + 0] = v.x;
                lA[r][q * 4 + 1] = v.y;
                lA[r][q * 4 + 2] = v.z;
                lA[r][q * 4 + 3] = v.w;
            }
        } else {
            for (int i = tid; i < ROWS * 32; i += NTHREADS) {
                int r = i >> 5, kk = i & 31;
                int row = blockRow + r, k = k0 + kk;
                float v = 0.f;
                if (row < n_nodes && k < K_DIM) v = X[(size_t)row * K_DIM + k];
                lA[r][kk] = v;
            }
        }
        // ---- stage B tile (32 x N_DIM) transposed into lB[col][k] ----
        if (fullK) {
            #pragma unroll
            for (int i = tid; i < 32 * (N_DIM / 4); i += NTHREADS) {
                int kk = i / (N_DIM / 4), c4 = i % (N_DIM / 4);
                float4 v = *(const float4*)&Wm[(size_t)(k0 + kk) * N_DIM + c4 * 4];
                lB[c4 * 4 + 0][kk] = v.x;
                lB[c4 * 4 + 1][kk] = v.y;
                lB[c4 * 4 + 2][kk] = v.z;
                lB[c4 * 4 + 3][kk] = v.w;
            }
        } else {
            for (int i = tid; i < 32 * N_DIM; i += NTHREADS) {
                int kk = i / N_DIM, c = i % N_DIM;
                int k  = k0 + kk;
                float v = 0.f;
                if (k < K_DIM) v = Wm[(size_t)k * N_DIM + c];
                lB[c][kk] = v;
            }
        }
        __syncthreads();

        // speculative prefetch of next A tile (global_prefetch_b8)
        if (ks + 1 < KSTEPS) {
            int row = blockRow + (tid % ROWS);
            int k = k0 + 32;
            if (row < n_nodes && k < K_DIM)
                __builtin_prefetch(&X[(size_t)row * K_DIM + k], 0, 0);
        }

        // ---- build f16 fragments per CDNA5 16-bit A/B VGPR layout ----
        // half pair in VGPR v covers K = (v>=4?16:0) + laneHi*8 + 2*(v&3) .. +1
        v16h a0, a1, b;
        #pragma unroll
        for (int v = 0; v < 8; ++v) {
            int kk = ((v >> 2) << 4) + (laneHi << 3) + ((v & 3) << 1);
            float2 fb = *(const float2*)&lB[(nt << 4) + laneLo][kk];
            b[2 * v]     = (_Float16)fb.x;
            b[2 * v + 1] = (_Float16)fb.y;
            float2 f0 = *(const float2*)&lA[rbase + laneLo][kk];
            a0[2 * v]     = (_Float16)f0.x;
            a0[2 * v + 1] = (_Float16)f0.y;
            float2 f1 = *(const float2*)&lA[rbase + 16 + laneLo][kk];
            a1[2 * v]     = (_Float16)f1.x;
            a1[2 * v + 1] = (_Float16)f1.y;
        }
        acc0 = __builtin_amdgcn_wmma_f32_16x16x32_f16(false, a0, false, b,
                                                      (short)0, acc0, false, false);
        acc1 = __builtin_amdgcn_wmma_f32_16x16x32_f16(false, a1, false, b,
                                                      (short)0, acc1, false, false);

        // ---- gate dot products: reuse staged A tile; branch-free.
        // lA is zero for k >= K_DIM, so clamping the sc/dkw index is exact.
        if (tid < ROWS) {
            #pragma unroll
            for (int kk = 0; kk < 32; ++kk) {
                int k = k0 + kk;
                k = (k < K_DIM) ? k : (K_DIM - 1);   // uniform clamp, no branch
                float xa = lA[tid][kk];
                gs = fmaf(xa, sc[k],  gs);
                gd = fmaf(xa, dkw[k], gd);
            }
        }
        __syncthreads();
    }

    // ---- store two 16x16 C tiles: VGPR i -> (M=i + laneHi*8, N=laneLo) ----
    const int col = (nt << 4) + laneLo;
    if (fullRows) {
        #pragma unroll
        for (int v = 0; v < 8; ++v) {
            int m = v + (laneHi << 3);
            size_t r0 = (size_t)(blockRow + rbase + m);
            XW[r0 * N_DIM + col]                 = acc0[v];
            XW[(r0 + 16) * N_DIM + col]          = acc1[v];
        }
    } else {
        #pragma unroll
        for (int v = 0; v < 8; ++v) {
            int m = v + (laneHi << 3);
            int row0 = blockRow + rbase + m;
            if (row0 < n_nodes)      XW[(size_t)row0 * N_DIM + col]        = acc0[v];
            if (row0 + 16 < n_nodes) XW[(size_t)(row0 + 16) * N_DIM + col] = acc1[v];
        }
    }

    if (tid < ROWS) {
        int row = blockRow + tid;
        if (row < n_nodes) {
            float sv = gs + sb[0];
            s_out[row]  = 1.f / (1.f + __expf(-sv));
            dk_out[row] = gd + dkb[0];
        }
    }
}

// ---------------------------------------------------------------------------
// SpMM scatter: one thread per (edge, feature). Lane index = feature ->
// fully coalesced gather of xW[src,:] and coalesced f32 atomics on agg[dst,:].
// ---------------------------------------------------------------------------
template<int H>
__global__ __launch_bounds__(256)
void spmm_kernel(const int* __restrict__ ei,   // [2, n_edges]
                 const float* __restrict__ ew, // [n_edges]
                 const float* __restrict__ feat,
                 float* __restrict__ out,
                 int n_edges)
{
    constexpr int PER = 256 / H;
    const int f = threadIdx.x % H;
    const int e = blockIdx.x * PER + threadIdx.x / H;
    if (e >= n_edges) return;
    const int src = ei[e];
    const int dst = ei[n_edges + e];
    const float w = ew[e];
    atomicAdd(out + (size_t)dst * H + f, w * feat[(size_t)src * H + f]);
}

// ---------------------------------------------------------------------------
// x_next = s*agg + (1-s)*agg_knn + GAMMA*Dk*xW
// ---------------------------------------------------------------------------
template<int H>
__global__ __launch_bounds__(256)
void combine_kernel(const float* __restrict__ xw,
                    const float* __restrict__ agg,
                    const float* __restrict__ aggk,
                    const float* __restrict__ s,
                    const float* __restrict__ dk,
                    float* __restrict__ out,
                    int n_nodes)
{
    int i = blockIdx.x * 256 + threadIdx.x;
    if (i >= n_nodes * H) return;
    int row = i / H;
    float sv = s[row], dv = dk[row];
    out[i] = sv * agg[i] + (1.f - sv) * aggk[i] + SIMP_GAMMA * dv * xw[i];
}

extern "C" void kernel_launch(void* const* d_in, const int* in_sizes, int n_in,
                              void* d_out, int out_size, void* d_ws, size_t ws_size,
                              hipStream_t stream)
{
    const float* x   = (const float*)d_in[0];
    const int*   ei  = (const int*)  d_in[1];
    const float* ew  = (const float*)d_in[2];
    const int*   kei = (const int*)  d_in[3];
    const float* kew = (const float*)d_in[4];
    const float* W0  = (const float*)d_in[5];
    const float* W1  = (const float*)d_in[6];
    const float* sc0 = (const float*)d_in[7];
    const float* sb0 = (const float*)d_in[8];
    const float* sc1 = (const float*)d_in[9];
    const float* sb1 = (const float*)d_in[10];
    const float* dk0 = (const float*)d_in[11];
    const float* db0 = (const float*)d_in[12];
    const float* dk1 = (const float*)d_in[13];
    const float* db1 = (const float*)d_in[14];

    constexpr int F = 500, H = 64, O = 16;
    const int E  = in_sizes[2];
    const int Ek = in_sizes[4];
    const int N  = in_sizes[0] / F;

    // workspace layout (floats): 258*N total (~52 MB for N=50000)
    float* ws   = (float*)d_ws;
    float* xw   = ws;                       // 64*N  (layer2 reuses first 16*N)
    float* agg  = ws + (size_t)64  * N;     // 64*N
    float* aggk = ws + (size_t)128 * N;     // 64*N
    float* x1   = ws + (size_t)192 * N;     // 64*N
    float* sg   = ws + (size_t)256 * N;     // N
    float* dg   = ws + (size_t)257 * N;     // N

    // ---------------- layer 1 ----------------
    hipMemsetAsync(agg,  0, (size_t)H * N * sizeof(float), stream);
    hipMemsetAsync(aggk, 0, (size_t)H * N * sizeof(float), stream);
    gemm_gate_wmma<F, H, 1, 4><<<(N + 31) / 32, 128, 0, stream>>>(
        x, W0, sc0, sb0, dk0, db0, xw, sg, dg, N);
    spmm_kernel<H><<<(E  + (256 / H) - 1) / (256 / H), 256, 0, stream>>>(ei,  ew,  xw, agg,  E);
    spmm_kernel<H><<<(Ek + (256 / H) - 1) / (256 / H), 256, 0, stream>>>(kei, kew, xw, aggk, Ek);
    combine_kernel<H><<<((unsigned)(N * H) + 255) / 256, 256, 0, stream>>>(
        xw, agg, aggk, sg, dg, x1, N);

    // ---------------- layer 2 ----------------
    hipMemsetAsync(agg,  0, (size_t)O * N * sizeof(float), stream);
    hipMemsetAsync(aggk, 0, (size_t)O * N * sizeof(float), stream);
    gemm_gate_wmma<H, O, 4, 1><<<(N + 127) / 128, 128, 0, stream>>>(
        x1, W1, sc1, sb1, dk1, db1, xw, sg, dg, N);
    spmm_kernel<O><<<(E  + (256 / O) - 1) / (256 / O), 256, 0, stream>>>(ei,  ew,  xw, agg,  E);
    spmm_kernel<O><<<(Ek + (256 / O) - 1) / (256 / O), 256, 0, stream>>>(kei, kew, xw, aggk, Ek);
    combine_kernel<O><<<((unsigned)(N * O) + 255) / 256, 256, 0, stream>>>(
        xw, agg, aggk, sg, dg, (float*)d_out, N);

    (void)n_in; (void)out_size; (void)ws_size;
}
